// SAModule_13348758356090
// MI455X (gfx1250) — compile-verified, matrix operand order checked by reading, not run on hardware
//
#include <hip/hip_runtime.h>
#include <cfloat>

#define N_PTS 16384
#define M_S   4096
#define KNN_K 16
#define E_EDGE (N_PTS * KNN_K)
#define EMBED 64

typedef __attribute__((ext_vector_type(16))) __bf16          v16bf;
typedef __attribute__((ext_vector_type(8)))  float           v8f;
typedef __attribute__((ext_vector_type(16))) unsigned short  v16us;
typedef __attribute__((ext_vector_type(8)))  unsigned short  v8us;

// ---------------- WMMA helpers (gfx1250, wave32) ----------------

static __device__ __forceinline__ v8f vzero8() {
    v8f z;
#pragma unroll
    for (int r = 0; r < 8; ++r) z[r] = 0.0f;
    return z;
}

static __device__ __forceinline__ v8f wmma_bf16(v16bf a, v16bf b, v8f c) {
    // D = A(16x32 bf16) * B(32x16 bf16) + C(16x16 f32)
    return __builtin_amdgcn_wmma_f32_16x16x32_bf16(false, a, false, b, (short)0, c,
                                                   false, false);
}

static __device__ __forceinline__ unsigned short f2bf(float f) {
    return __builtin_bit_cast(unsigned short, (__bf16)f);
}

// Assemble a 16-element bf16 fragment from two contiguous 16B runs at p and p+32B.
// (ISA 16-bit A layout: lane's K elements are {koff+0..7} and {16+koff+0..7}.)
static __device__ __forceinline__ v16bf a_frag_ptr(const unsigned short* p) {
    v8us lo = *(const v8us*)p;
    v8us hi = *(const v8us*)(p + 16);
    v16us c;
#pragma unroll
    for (int e = 0; e < 8; ++e) { c[e] = lo[e]; c[8 + e] = hi[e]; }
    return __builtin_bit_cast(v16bf, c);
}

// A-fragment from a row-major bf16 matrix in global memory, per-lane row gather.
static __device__ __forceinline__ v16bf a_frag_bfg(const unsigned short* base, int row,
                                                   int ld, int k0, int lane) {
    int koff = (lane >> 4) << 3;
    return a_frag_ptr(base + (size_t)row * ld + k0 + koff);
}

// A-fragment from a per-wave LDS bf16 tile [16][ldk] (m = lane&15).
static __device__ __forceinline__ v16bf a_frag_bfl(const unsigned short* t, int ldk,
                                                   int k0, int lane) {
    int mm = lane & 15;
    int koff = (lane >> 4) << 3;
    return a_frag_ptr(t + mm * ldk + k0 + koff);
}

// B-fragment from pre-packed weights: fragment (kb,nt) is a contiguous 32x16 bf16
// block; each lane loads its own 32 contiguous bytes (one v16us -> 2x b128).
static __device__ __forceinline__ v16bf b_frag_pk(const unsigned short* Wpk, int kb,
                                                  int nt, int NT, int lane) {
    const v16us* p =
        (const v16us*)(Wpk + ((((size_t)(kb * NT + nt)) * 32 + lane) << 4));
    return __builtin_bit_cast(v16bf, *p);
}

// Store one C tile column-set to a bf16 LDS tile with bias (+optional relu).
static __device__ __forceinline__ void store_tile_bf(unsigned short* t, int ldk, int col,
                                                     int hi, v8f acc, float bias,
                                                     bool relu) {
#pragma unroll
    for (int r = 0; r < 8; ++r) {
        float v = acc[r] + bias;
        if (relu) v = fmaxf(v, 0.0f);
        t[(r + 8 * hi) * ldk + col] = f2bf(v);
    }
}

// Deterministic float atomic-max via integer punning (positive: signed max,
// negative: unsigned min). Init value must be -inf bits (0xFF800000).
static __device__ __forceinline__ void atomicMaxF(float* addr, float v) {
    if (v >= 0.0f)
        atomicMax((int*)addr, __float_as_int(v));
    else
        atomicMin((unsigned int*)addr, __float_as_uint(v));
}

// ---------------- Kernel 1: weight prep (packed bf16 fragments) + aggr init ----------------

// Write element i of a packed-fragment weight image. Layout:
//   frag f = kb*NT + nt ; within frag: lane L (32) x element e (16), contiguous.
//   value = W[k][n], n = nt*16 + (L&15), k = kb*32 + (e<8 ? koff+e : 16+koff+e-8),
//   koff = (L>>4)*8 ; zero-padded outside [K_real, N_real].
static __device__ __forceinline__ void pack_weight(const float* __restrict__ W,
                                                   unsigned short* __restrict__ dst,
                                                   int i, int K_real, int N_real,
                                                   int NT) {
    int e = i & 15;
    int L = (i >> 4) & 31;
    int f = i >> 9;
    int nt = f % NT;
    int kb = f / NT;
    int n = nt * 16 + (L & 15);
    int koff = (L >> 4) << 3;
    int k = kb * 32 + (e < 8 ? koff + e : 16 + koff + (e - 8));
    float v = (k < K_real && n < N_real) ? W[k * N_real + n] : 0.0f;
    dst[i] = f2bf(v);
}

__global__ void prep_kernel(const float* __restrict__ h_w1, const float* __restrict__ h_w2,
                            const float* __restrict__ f_w1, const float* __restrict__ f_w2,
                            const float* __restrict__ g_w1, const float* __restrict__ g_w2,
                            unsigned short* __restrict__ wh1, unsigned short* __restrict__ wh2,
                            unsigned short* __restrict__ wf1, unsigned short* __restrict__ wf2,
                            unsigned short* __restrict__ wg1, unsigned short* __restrict__ wg2,
                            float* __restrict__ aggr) {
    int i = blockIdx.x * 256 + threadIdx.x;
    if (i < M_S * EMBED) aggr[i] = __uint_as_float(0xFF800000u);  // -inf
    if (i < 4096) pack_weight(h_w1, wh1, i, 64, 64, 4);   // [64][64]
    if (i < 1024) pack_weight(h_w2, wh2, i, 64, 3, 1);    // [64][3] -> N pad 16
    if (i < 6144) pack_weight(f_w1, wf1, i, 67, 64, 4);   // [67][64] -> K pad 96
    if (i < 4096) pack_weight(f_w2, wf2, i, 64, 64, 4);   // [64][64]
    if (i < 8192) pack_weight(g_w1, wg1, i, 128, 64, 4);  // [128][64]
    if (i < 4096) pack_weight(g_w2, wg2, i, 64, 64, 4);   // [64][64]
}

// ---------------- Kernel 2: farthest point sampling (single WG, LDS-resident) ----------------

__global__ void __launch_bounds__(1024) fps_kernel(const float* __restrict__ pos,
                                                   int* __restrict__ idx_out) {
    extern __shared__ float smem[];           // 4*N floats = 256KB (LDS is 320KB/WGP)
    float* smin = smem;
    float* sx = smem + N_PTS;
    float* sy = smem + 2 * N_PTS;
    float* sz = smem + 3 * N_PTS;
    __shared__ float redv[32];
    __shared__ int redi[32];
    __shared__ int s_last;

    int tid = threadIdx.x;
    for (int i = tid; i < N_PTS; i += 1024) {
        sx[i] = pos[3 * i];
        sy[i] = pos[3 * i + 1];
        sz[i] = pos[3 * i + 2];
        smin[i] = FLT_MAX;
    }
    if (tid == 0) {
        idx_out[0] = 0;
        s_last = 0;
    }
    __syncthreads();

    int lane = tid & 31, warp = tid >> 5;
    for (int s = 1; s < M_S; ++s) {
        int last = s_last;
        float px = sx[last], py = sy[last], pz = sz[last];
        float bestv = -FLT_MAX;
        int besti = 0;
        for (int i = tid; i < N_PTS; i += 1024) {
            float dx = sx[i] - px, dy = sy[i] - py, dz = sz[i] - pz;
            float d = dx * dx + dy * dy + dz * dz;
            float nm = fminf(smin[i], d);
            smin[i] = nm;
            if (nm > bestv) { bestv = nm; besti = i; }   // strict > keeps first index
        }
#pragma unroll
        for (int off = 16; off; off >>= 1) {
            float ov = __shfl_down(bestv, off, 32);
            int oi = __shfl_down(besti, off, 32);
            if (ov > bestv || (ov == bestv && oi < besti)) { bestv = ov; besti = oi; }
        }
        if (lane == 0) { redv[warp] = bestv; redi[warp] = besti; }
        __syncthreads();
        if (warp == 0) {
            bestv = redv[lane];
            besti = redi[lane];
#pragma unroll
            for (int off = 16; off; off >>= 1) {
                float ov = __shfl_down(bestv, off, 32);
                int oi = __shfl_down(besti, off, 32);
                if (ov > bestv || (ov == bestv && oi < besti)) { bestv = ov; besti = oi; }
            }
            if (lane == 0) {
                idx_out[s] = besti;
                s_last = besti;
            }
        }
        __syncthreads();
    }
}

// ---------------- Kernel 3: gather sampled rows (f32 + bf16 images) ----------------

__global__ void gather_kernel(const float* __restrict__ x, const float* __restrict__ pos,
                              const int* __restrict__ idx, float* __restrict__ x_s,
                              unsigned short* __restrict__ x_s_bf,
                              float* __restrict__ pos_s, float* __restrict__ out_pos_s) {
    int i = blockIdx.x * 256 + threadIdx.x;
    if (i < M_S * EMBED) {
        float v = x[(size_t)idx[i >> 6] * EMBED + (i & 63)];
        x_s[i] = v;
        x_s_bf[i] = f2bf(v);
    }
    if (i < M_S * 3) {
        float v = pos[(size_t)idx[i / 3] * 3 + (i % 3)];
        pos_s[i] = v;
        out_pos_s[i] = v;
    }
}

// ---------------- Kernel 4: KNN (top-16 insertion network in registers) ----------------

__global__ void __launch_bounds__(128) knn_kernel(const float* __restrict__ pos,
                                                  const float* __restrict__ pos_s,
                                                  int* __restrict__ tgt,
                                                  float* __restrict__ out_assign) {
    __shared__ float sxm[M_S], sym[M_S], szm[M_S];   // 48KB: all sampled positions
    for (int i = threadIdx.x; i < M_S; i += 128) {
        sxm[i] = pos_s[3 * i];
        sym[i] = pos_s[3 * i + 1];
        szm[i] = pos_s[3 * i + 2];
    }
    __syncthreads();

    int q = blockIdx.x * 128 + threadIdx.x;
    float px = pos[3 * q], py = pos[3 * q + 1], pz = pos[3 * q + 2];
    float bd[KNN_K];
    int bi[KNN_K];
#pragma unroll
    for (int t = 0; t < KNN_K; ++t) { bd[t] = FLT_MAX; bi[t] = 0; }

    for (int j = 0; j < M_S; ++j) {
        float dx = px - sxm[j], dy = py - sym[j], dz = pz - szm[j];
        float d = dx * dx + dy * dy + dz * dz;
        if (d < bd[KNN_K - 1]) {
            float cd = d;
            int ci = j;
#pragma unroll
            for (int t = 0; t < KNN_K; ++t) {   // stable insertion (ties keep lower j)
                if (cd < bd[t]) {
                    float tv = bd[t]; int ti = bi[t];
                    bd[t] = cd; bi[t] = ci;
                    cd = tv; ci = ti;
                }
            }
        }
    }
#pragma unroll
    for (int r = 0; r < KNN_K; ++r) {
        tgt[q * KNN_K + r] = bi[r];
        out_assign[q * KNN_K + r] = (float)q;               // src row of assign_index
        out_assign[E_EDGE + q * KNN_K + r] = (float)bi[r];  // tgt row
    }
}

// ---------------- Kernel 5: fused edge MLPs (h then f) + atomic max aggregation ----------------
// One wave = one query point = 16 edges (src[e]=e/16 so an aligned tile shares its query).

__global__ void __launch_bounds__(128) edge_kernel(
    const float* __restrict__ x, const float* __restrict__ pos,
    const unsigned short* __restrict__ x_s_bf, const float* __restrict__ pos_s,
    const int* __restrict__ tgt,
    const float* __restrict__ h_b1, const float* __restrict__ h_b2,
    const float* __restrict__ f_b1, const float* __restrict__ f_b2,
    const unsigned short* __restrict__ wh1, const unsigned short* __restrict__ wh2,
    const unsigned short* __restrict__ wf1, const unsigned short* __restrict__ wf2,
    float* __restrict__ aggr) {
    __shared__ unsigned short sbuf[4][16 * 96];   // per-wave bf16 staging (12KB/block)
    int lane = threadIdx.x & 31;
    int wave = threadIdx.x >> 5;
    int qi = blockIdx.x * 4 + wave;      // query index (tile = its 16 edges)
    unsigned short* buf = sbuf[wave];
    int m = lane & 15;
    int hi = lane >> 4;
    int e_base = qi * KNN_K;
    int rowA = tgt[e_base + m];          // x_s row feeding this lane's A row

    __builtin_prefetch(wf1, 0, 1);

    // ---- h1 = relu(x_i @ h_w1 + h_b1)  [16,64], A gathered from bf16 x_s ----
    v16bf a0 = a_frag_bfg(x_s_bf, rowA, EMBED, 0, lane);
    v16bf a1 = a_frag_bfg(x_s_bf, rowA, EMBED, 32, lane);
#pragma unroll
    for (int nt = 0; nt < 4; ++nt) {
        v8f acc = vzero8();
        acc = wmma_bf16(a0, b_frag_pk(wh1, 0, nt, 4, lane), acc);
        acc = wmma_bf16(a1, b_frag_pk(wh1, 1, nt, 4, lane), acc);
        int col = nt * 16 + m;
        store_tile_bf(buf, 64, col, hi, acc, h_b1[col], true);
    }
    __syncthreads();

    // ---- delta = h1 @ h_w2 + h_b2  [16,3] (N padded to 16, cols>=3 zero-weight) ----
    v16bf ha0 = a_frag_bfl(buf, 64, 0, lane);
    v16bf ha1 = a_frag_bfl(buf, 64, 32, lane);
    v8f dacc = vzero8();
    dacc = wmma_bf16(ha0, b_frag_pk(wh2, 0, 0, 1, lane), dacc);
    dacc = wmma_bf16(ha1, b_frag_pk(wh2, 1, 0, 1, lane), dacc);
    __syncthreads();

    // ---- build in_f = [distance+delta | x_j | 0-pad] as bf16 [16][96] in LDS ----
    for (int p = lane; p < 16 * 96; p += 32) {
        int cc = p % 96;
        if (cc >= 3)
            buf[p] = (cc < 67) ? f2bf(x[(size_t)qi * EMBED + (cc - 3)]) : (unsigned short)0;
    }
    if (m < 3) {   // lanes 0-2 / 16-18 own config column m
        float pxc = pos[qi * 3 + m];
        float b2 = h_b2[m];
#pragma unroll
        for (int r = 0; r < 8; ++r) {
            int row = r + 8 * hi;
            int trow = tgt[e_base + row];
            float dist = pxc - pos_s[trow * 3 + m];
            buf[row * 96 + m] = f2bf(dist + dacc[r] + b2);
        }
    }
    __syncthreads();

    // ---- f1 = relu(in_f @ f_w1 + f_b1)  [16,64], K padded 67->96 ----
    v16bf fa0 = a_frag_bfl(buf, 96, 0, lane);
    v16bf fa1 = a_frag_bfl(buf, 96, 32, lane);
    v16bf fa2 = a_frag_bfl(buf, 96, 64, lane);
    __syncthreads();
#pragma unroll
    for (int nt = 0; nt < 4; ++nt) {
        v8f acc = vzero8();
        acc = wmma_bf16(fa0, b_frag_pk(wf1, 0, nt, 4, lane), acc);
        acc = wmma_bf16(fa1, b_frag_pk(wf1, 1, nt, 4, lane), acc);
        acc = wmma_bf16(fa2, b_frag_pk(wf1, 2, nt, 4, lane), acc);
        int col = nt * 16 + m;
        store_tile_bf(buf, 64, col, hi, acc, f_b1[col], true);
    }
    __syncthreads();

    // ---- msg = f1 @ f_w2 + f_b2, then segment-max into aggr[tgt] ----
    v16bf ma0 = a_frag_bfl(buf, 64, 0, lane);
    v16bf ma1 = a_frag_bfl(buf, 64, 32, lane);
#pragma unroll
    for (int nt = 0; nt < 4; ++nt) {
        v8f acc = vzero8();
        acc = wmma_bf16(ma0, b_frag_pk(wf2, 0, nt, 4, lane), acc);
        acc = wmma_bf16(ma1, b_frag_pk(wf2, 1, nt, 4, lane), acc);
        int col = nt * 16 + m;
        float bv = f_b2[col];
#pragma unroll
        for (int r = 0; r < 8; ++r) {
            int row = r + 8 * hi;
            int trow = tgt[e_base + row];
            atomicMaxF(&aggr[(size_t)trow * EMBED + col], acc[r] + bv);
        }
    }
}

// ---------------- Kernel 6: g MLP over sampled nodes + residual ----------------

__global__ void __launch_bounds__(128) out_kernel(
    const float* __restrict__ x_s, const float* __restrict__ aggr,
    const float* __restrict__ g_b1, const float* __restrict__ g_b2,
    const unsigned short* __restrict__ wg1, const unsigned short* __restrict__ wg2,
    float* __restrict__ out_xnew) {
    __shared__ unsigned short sbuf[4][16 * 128];   // bf16 staging (16KB/block)
    int lane = threadIdx.x & 31;
    int wave = threadIdx.x >> 5;
    int tile = blockIdx.x * 4 + wave;     // 256 tiles of 16 sampled rows
    unsigned short* buf = sbuf[wave];
    int m = lane & 15;
    int hi = lane >> 4;
    int base_row = tile * 16;

    // in_g = [x_s | finite(aggr)]  bf16 [16][128]
    for (int p = lane; p < 16 * 128; p += 32) {
        int rr = p >> 7, cc = p & 127;
        int row = base_row + rr;
        float v;
        if (cc < 64) {
            v = x_s[(size_t)row * EMBED + cc];
        } else {
            float a = aggr[(size_t)row * EMBED + (cc - 64)];
            unsigned u = __float_as_uint(a);
            v = ((u & 0x7F800000u) == 0x7F800000u) ? 0.0f : a;  // where(isfinite, a, 0)
        }
        buf[p] = f2bf(v);
    }
    __syncthreads();

    v16bf ga0 = a_frag_bfl(buf, 128, 0, lane);
    v16bf ga1 = a_frag_bfl(buf, 128, 32, lane);
    v16bf ga2 = a_frag_bfl(buf, 128, 64, lane);
    v16bf ga3 = a_frag_bfl(buf, 128, 96, lane);
    __syncthreads();

    // g1 = relu(in_g @ g_w1 + g_b1)
#pragma unroll
    for (int nt = 0; nt < 4; ++nt) {
        v8f acc = vzero8();
        acc = wmma_bf16(ga0, b_frag_pk(wg1, 0, nt, 4, lane), acc);
        acc = wmma_bf16(ga1, b_frag_pk(wg1, 1, nt, 4, lane), acc);
        acc = wmma_bf16(ga2, b_frag_pk(wg1, 2, nt, 4, lane), acc);
        acc = wmma_bf16(ga3, b_frag_pk(wg1, 3, nt, 4, lane), acc);
        int col = nt * 16 + m;
        store_tile_bf(buf, 64, col, hi, acc, g_b1[col], true);
    }
    __syncthreads();

    // x_new = x_s + (g1 @ g_w2 + g_b2)   (residual kept in f32)
    v16bf oa0 = a_frag_bfl(buf, 64, 0, lane);
    v16bf oa1 = a_frag_bfl(buf, 64, 32, lane);
#pragma unroll
    for (int nt = 0; nt < 4; ++nt) {
        v8f acc = vzero8();
        acc = wmma_bf16(oa0, b_frag_pk(wg2, 0, nt, 4, lane), acc);
        acc = wmma_bf16(oa1, b_frag_pk(wg2, 1, nt, 4, lane), acc);
        int col = nt * 16 + m;
        float bv = g_b2[col];
#pragma unroll
        for (int r = 0; r < 8; ++r) {
            int grow = base_row + r + 8 * hi;
            out_xnew[(size_t)grow * EMBED + col] =
                acc[r] + bv + x_s[(size_t)grow * EMBED + col];
        }
    }
}

// ---------------- Launch ----------------

extern "C" void kernel_launch(void* const* d_in, const int* in_sizes, int n_in,
                              void* d_out, int out_size, void* d_ws, size_t ws_size,
                              hipStream_t stream) {
    const float* x    = (const float*)d_in[0];
    const float* pos  = (const float*)d_in[1];
    // d_in[2] = edge_index (int64) is stored-but-unused by the reference math
    const float* h_w1 = (const float*)d_in[3];
    const float* h_b1 = (const float*)d_in[4];
    const float* h_w2 = (const float*)d_in[5];
    const float* h_b2 = (const float*)d_in[6];
    const float* f_w1 = (const float*)d_in[7];
    const float* f_b1 = (const float*)d_in[8];
    const float* f_w2 = (const float*)d_in[9];
    const float* f_b2 = (const float*)d_in[10];
    const float* g_w1 = (const float*)d_in[11];
    const float* g_b1 = (const float*)d_in[12];
    const float* g_w2 = (const float*)d_in[13];
    const float* g_b2 = (const float*)d_in[14];

    // Workspace layout (~3.7 MB, all chunks 256B-aligned)
    char* ws = (char*)d_ws;
    int* idx = (int*)(ws);                                    // 16384 B
    int* tgt = (int*)(ws + 16384);                            // 1048576 B
    float* poss = (float*)(ws + 1064960);                     // 49152 B
    float* xs = (float*)(ws + 1114112);                       // 1048576 B (f32)
    unsigned short* xsbf = (unsigned short*)(ws + 2162688);   // 524288 B (bf16)
    float* aggr = (float*)(ws + 2686976);                     // 1048576 B
    unsigned short* wh1 = (unsigned short*)(ws + 3735552);    // packed weights
    unsigned short* wh2 = wh1 + 4096;
    unsigned short* wf1 = wh2 + 1024;
    unsigned short* wf2 = wf1 + 6144;
    unsigned short* wg1 = wf2 + 4096;
    unsigned short* wg2 = wg1 + 8192;

    float* out_xnew = (float*)d_out;                 // [M,64]
    float* out_poss = out_xnew + M_S * EMBED;        // [M,3]
    float* out_assign = out_poss + M_S * 3;          // [2,E] as floats

    prep_kernel<<<1024, 256, 0, stream>>>(h_w1, h_w2, f_w1, f_w2, g_w1, g_w2,
                                          wh1, wh2, wf1, wf2, wg1, wg2, aggr);
    fps_kernel<<<1, 1024, (size_t)N_PTS * 4 * sizeof(float), stream>>>(pos, idx);
    gather_kernel<<<1024, 256, 0, stream>>>(x, pos, idx, xs, xsbf, poss, out_poss);
    knn_kernel<<<N_PTS / 128, 128, 0, stream>>>(pos, poss, tgt, out_assign);
    edge_kernel<<<N_PTS / 4, 128, 0, stream>>>(x, pos, xsbf, poss, tgt,
                                               h_b1, h_b2, f_b1, f_b2,
                                               wh1, wh2, wf1, wf2, aggr);
    out_kernel<<<M_S / 16 / 4, 128, 0, stream>>>(xs, aggr, g_b1, g_b2, wg1, wg2,
                                                 out_xnew);
}
